// VQVAE_60069412602505
// MI455X (gfx1250) — matrix operand order repaired and checked by reading, not compiled, not run on hardware
//
#include <hip/hip_runtime.h>
#include <hip/hip_bf16.h>

// ---------------------------------------------------------------------------
// VQ-VAE forward on MI455X (gfx1250, wave32).
// All GEMMs use v_wmma_f32_16x16x32_f16 (f16 in, f32 accumulate).
// Argmin over the 8192-entry codebook is a streaming register argmax of
// (lat.emb - 0.5*|emb|^2); the [N,K] distance matrix is never materialized.
// Codebook tiles are double-buffered through LDS with async-to-LDS DMA.
// Forward identities used: z == q,  vq_loss == (1+BETA)*mean((q-lat)^2).
// ---------------------------------------------------------------------------

typedef _Float16 f16;
typedef __attribute__((ext_vector_type(16))) _Float16 v16h;
typedef __attribute__((ext_vector_type(8)))  _Float16 v8h;
typedef __attribute__((ext_vector_type(8)))  float    v8f;
typedef __attribute__((ext_vector_type(4)))  float    v4f;
typedef __attribute__((ext_vector_type(4)))  int      v4i;

// pointer types matching the async-to-LDS builtin's signature (v4i*1, v4i*3)
typedef __attribute__((address_space(1))) v4i gv4i;
typedef __attribute__((address_space(3))) v4i lv4i;

#define VQN      32768
#define VQ_INDIM 128
#define VQ_LAT   64
#define VQ_K     8192
#define VQ_BETA  1.25f
#define VQ_STAGE 128           // codebook rows per LDS stage (16 KB), 2 buffers

#if __has_builtin(__builtin_amdgcn_global_load_async_to_lds_b128)
#define HAVE_ASYNC_LDS 1
#endif

__device__ __forceinline__ void wait_async_lds() {
#if __has_builtin(__builtin_amdgcn_s_wait_asynccnt)
    __builtin_amdgcn_s_wait_asynccnt(0);
#else
    asm volatile("s_wait_asynccnt 0x0" ::: "memory");
#endif
}

// ---------------- fragment loaders (CDNA5 16-bit WMMA layouts) -------------
// A 16x32 f16 (MxK): lane l (r=l&15,g=l>>4) row M=r.
//   elements 0..7  -> K = k0 + 8*g + e
//   elements 8..15 -> K = k0 + 16 + 8*g + (e-8)
__device__ __forceinline__ v16h load_a_frag(const f16* __restrict__ A, int lda,
                                            int row, int k0, int g) {
    const f16* p = A + (size_t)row * lda + k0 + 8 * g;
    union { v16h v; v8h h[2]; } u;
    u.h[0] = *(const v8h*)(p);
    u.h[1] = *(const v8h*)(p + 16);
    return u.v;
}

// B 32x16 f16 (KxN), sourced from row-major W[Nout x K] (B[k][n] = W[n][k]):
// lane l: column n = n0 + (l&15); elements 0..7 -> K = k0 + 8*g + e, 8..15 -> K = k0 + 16 + 8*g + (e-8).
__device__ __forceinline__ v16h load_b_frag(const f16* __restrict__ W, int ldw,
                                            int col, int k0, int g) {
    const f16* p = W + (size_t)col * ldw + k0 + 8 * g;
    union { v16h v; v8h h[2]; } u;
    u.h[0] = *(const v8h*)(p);
    u.h[1] = *(const v8h*)(p + 16);
    return u.v;
}

__device__ __forceinline__ v8f wmma_f16(v16h a, v16h b, v8f c) {
    return __builtin_amdgcn_wmma_f32_16x16x32_f16(
        /*neg_a=*/false, a, /*neg_b=*/false, b,
        /*c_mod=*/(short)0, c, /*reuse_a=*/false, /*reuse_b=*/false);
}

// ---------------- conversion / precompute kernels --------------------------
__global__ void f32_to_f16_kernel(const float* __restrict__ in,
                                  f16* __restrict__ out, int n) {
    int i = blockIdx.x * blockDim.x + threadIdx.x;
    if (i < n) out[i] = (f16)in[i];
}

__global__ void emb_halfnorm_kernel(const float* __restrict__ emb,
                                    float* __restrict__ nrm, int K) {
    int k = blockIdx.x * blockDim.x + threadIdx.x;
    if (k >= K) return;
    const float* e = emb + (size_t)k * VQ_LAT;
    float s = 0.f;
#pragma unroll 4
    for (int d = 0; d < VQ_LAT; d += 4) {
        v4f v = *(const v4f*)(e + d);
        s += v[0] * v[0] + v[1] * v[1] + v[2] * v[2] + v[3] * v[3];
    }
    nrm[k] = 0.5f * s;
}

// zero one_hot region (b128 stores) + zero vq_loss slot
__global__ void fill_zero_kernel(float* __restrict__ p, size_t n4,
                                 float* __restrict__ loss_slot) {
    size_t i = (size_t)blockIdx.x * blockDim.x + threadIdx.x;
    size_t stride = (size_t)gridDim.x * blockDim.x;
    v4f z = {0.f, 0.f, 0.f, 0.f};
    for (; i < n4; i += stride) ((v4f*)p)[i] = z;
    if (blockIdx.x == 0 && threadIdx.x == 0) *loss_slot = 0.f;
}

// ---------------- generic WMMA GEMM: out = act(A * W^T + b) ----------------
// A: [M x Kd] f16 row-major. W: [Nout x Kd] f16 row-major. bias: [Nout] f32.
// One 16x16 output tile per wave; 4 waves per block cover 64 output columns.
__global__ __launch_bounds__(128)
void gemm_wmma_kernel(const f16* __restrict__ A, const f16* __restrict__ W,
                      const float* __restrict__ bias,
                      float* __restrict__ outF, f16* __restrict__ outH,
                      int M, int Kd, int Nout, int relu) {
    const int wave = threadIdx.x >> 5;
    const int lane = threadIdx.x & 31;
    const int g = lane >> 4, ln = lane & 15;
    const int m0 = blockIdx.x * 16;
    const int n0 = (blockIdx.y * 4 + wave) * 16;
    if (n0 >= Nout || m0 >= M) return;

    v8f c = {};
    const int arow = m0 + ln;
    const int bcol = n0 + ln;
    for (int k0 = 0; k0 < Kd; k0 += 32) {
        v16h a = load_a_frag(A, Kd, arow, k0, g);
        v16h b = load_b_frag(W, Kd, bcol, k0, g);
        c = wmma_f16(a, b, c);
    }

    const float bv = bias[bcol];
    const int rbase = m0 + 8 * g;          // C/D: VGPR r -> row r (+8 for lanes>=16)
#pragma unroll
    for (int r = 0; r < 8; ++r) {
        float v = c[r] + bv;
        if (relu) v = fmaxf(v, 0.f);
        const size_t o = (size_t)(rbase + r) * Nout + bcol;
        if (outF) outF[o] = v;
        if (outH) outH[o] = (f16)v;
    }
}

// ---------------- VQ argmax kernel -----------------------------------------
// Per 16-row tile of lat, stream all K codes through a double-buffered LDS
// pipeline fed by async-to-LDS DMA; score via WMMA; running (max score,
// first index) in registers; shfl_xor reduction at the end.
__global__ __launch_bounds__(256)
void vq_argmax_kernel(const f16* __restrict__ lat16,
                      const f16* __restrict__ emb16,
                      const float* __restrict__ embnorm,
                      int* __restrict__ inds, int N, int K) {
    __shared__ __align__(16) f16   sEmb[2 * VQ_STAGE * VQ_LAT];   // 32 KB
    __shared__ float sNorm[2][VQ_STAGE];

    const int wave = threadIdx.x >> 5;
    const int lane = threadIdx.x & 31;
    const int g = lane >> 4, ln = lane & 15;
    const int m0 = (blockIdx.x * 8 + wave) * 16;
    const int nstage = K / VQ_STAGE;

    auto issue_stage = [&](int s, int buf) {
        const v8h* src = (const v8h*)(emb16 + (size_t)s * VQ_STAGE * VQ_LAT);
        v8h* dst = (v8h*)(sEmb + buf * VQ_STAGE * VQ_LAT);
#if defined(HAVE_ASYNC_LDS)
        for (int i = threadIdx.x; i < VQ_STAGE * (VQ_LAT / 8); i += blockDim.x)
            __builtin_amdgcn_global_load_async_to_lds_b128(
                (gv4i*)(src + i), (lv4i*)(dst + i), /*offset=*/0, /*cpol=*/0);
#else
        for (int i = threadIdx.x; i < VQ_STAGE * (VQ_LAT / 8); i += blockDim.x)
            dst[i] = src[i];
#endif
        for (int i = threadIdx.x; i < VQ_STAGE; i += blockDim.x)
            sNorm[buf][i] = embnorm[s * VQ_STAGE + i];
    };

    // A fragments (lat rows) reused across all 8192 codes
    const v16h a0 = load_a_frag(lat16, VQ_LAT, m0 + ln, 0, g);
    const v16h a1 = load_a_frag(lat16, VQ_LAT, m0 + ln, 32, g);

    float best[8];
    int   bidx[8];
#pragma unroll
    for (int r = 0; r < 8; ++r) { best[r] = -3.4e38f; bidx[r] = 0; }

    issue_stage(0, 0);
    for (int s = 0; s < nstage; ++s) {
        const int cur = s & 1;
        wait_async_lds();        // stage s resident in sEmb[cur]
        __syncthreads();         // ...for every wave; stage s-1 reads all done
        if (s + 1 < nstage) issue_stage(s + 1, cur ^ 1);   // overlap DMA w/ WMMA

        const f16* eb = sEmb + cur * VQ_STAGE * VQ_LAT;
        for (int t = 0; t < VQ_STAGE; t += 16) {
            v16h b0 = load_b_frag(eb, VQ_LAT, t + ln, 0, g);
            v16h b1 = load_b_frag(eb, VQ_LAT, t + ln, 32, g);
            v8f c = {};
            c = wmma_f16(a0, b0, c);
            c = wmma_f16(a1, b1, c);
            const float nv = sNorm[cur][t + ln];
            const int gidx = s * VQ_STAGE + t + ln;
#pragma unroll
            for (int r = 0; r < 8; ++r) {
                const float sc = c[r] - nv;     // dot - 0.5*|emb|^2
                // per-lane indices increase monotonically, so strict '>' keeps
                // the first (smallest-index) maximum: no tie-break needed here
                if (sc > best[r]) { best[r] = sc; bidx[r] = gidx; }
            }
        }
    }

    // reduce across the 16 lanes holding the same rows (xor within half-wave)
#pragma unroll
    for (int off = 8; off > 0; off >>= 1) {
#pragma unroll
        for (int r = 0; r < 8; ++r) {
            float ob = __shfl_xor(best[r], off, 32);
            int   oi = __shfl_xor(bidx[r], off, 32);
            if (ob > best[r] || (ob == best[r] && oi < bidx[r])) {
                best[r] = ob; bidx[r] = oi;
            }
        }
    }
    if (ln == 0) {
#pragma unroll
        for (int r = 0; r < 8; ++r) inds[m0 + 8 * g + r] = bidx[r];
    }
}

// ---------------- post-VQ: one_hot scatter, q gather, fused loss -----------
__global__ __launch_bounds__(256)
void vq_post_kernel(const int* __restrict__ inds,
                    const float* __restrict__ emb, const f16* __restrict__ emb16,
                    const float* __restrict__ lat32,
                    float* __restrict__ q32, f16* __restrict__ q16,
                    float* __restrict__ onehot, float* __restrict__ loss_slot,
                    int N, int K) {
    const int n = blockIdx.x * blockDim.x + threadIdx.x;
    float partial = 0.f;
    if (n < N) {
        const int ind = inds[n];
        onehot[(size_t)n * K + ind] = 1.0f;
        const float* e  = emb   + (size_t)ind * VQ_LAT;
        const f16*   eh = emb16 + (size_t)ind * VQ_LAT;
        const float* l  = lat32 + (size_t)n * VQ_LAT;
        float* q  = q32 + (size_t)n * VQ_LAT;
        f16*   qh = q16 + (size_t)n * VQ_LAT;
#pragma unroll 4
        for (int d = 0; d < VQ_LAT; d += 8) {
            v4f e0 = *(const v4f*)(e + d);
            v4f e1 = *(const v4f*)(e + d + 4);
            v4f l0 = *(const v4f*)(l + d);
            v4f l1 = *(const v4f*)(l + d + 4);
            *(v4f*)(q + d)     = e0;
            *(v4f*)(q + d + 4) = e1;
            *(v8h*)(qh + d)    = *(const v8h*)(eh + d);   // q16 = emb16[ind]
            v4f d0 = e0 - l0, d1 = e1 - l1;
            partial += d0[0]*d0[0] + d0[1]*d0[1] + d0[2]*d0[2] + d0[3]*d0[3]
                     + d1[0]*d1[0] + d1[1]*d1[1] + d1[2]*d1[2] + d1[3]*d1[3];
        }
    }
#pragma unroll
    for (int off = 16; off > 0; off >>= 1) partial += __shfl_xor(partial, off, 32);
    if ((threadIdx.x & 31) == 0) {
        const float scale = (1.0f + VQ_BETA) / ((float)VQN * (float)VQ_LAT);
        atomicAdd(loss_slot, partial * scale);
    }
}

// ---------------------------------------------------------------------------
extern "C" void kernel_launch(void* const* d_in, const int* in_sizes, int n_in,
                              void* d_out, int out_size, void* d_ws, size_t ws_size,
                              hipStream_t stream) {
    (void)in_sizes; (void)n_in; (void)out_size; (void)ws_size;
    const int N = VQN, K = VQ_K;

    // inputs (setup_inputs order)
    const float* x      = (const float*)d_in[0];
    const float* eW1    = (const float*)d_in[1];
    const float* eb1    = (const float*)d_in[2];
    const float* eW2    = (const float*)d_in[3];
    const float* eb2    = (const float*)d_in[4];
    const float* eW3    = (const float*)d_in[5];
    const float* eb3    = (const float*)d_in[6];
    const float* emb    = (const float*)d_in[7];
    const float* dW1    = (const float*)d_in[8];
    const float* db1    = (const float*)d_in[9];
    const float* dW2    = (const float*)d_in[10];
    const float* db2    = (const float*)d_in[11];
    const float* dWmu   = (const float*)d_in[12];
    const float* dbmu   = (const float*)d_in[13];
    const float* dWlv   = (const float*)d_in[14];
    const float* dblv   = (const float*)d_in[15];

    // outputs (flat concat): x_hat[N*128], x_hat_var[N*128], one_hot[N*K], vq_loss[1]
    float* out_mu     = (float*)d_out;
    float* out_lv     = out_mu + (size_t)N * VQ_INDIM;
    float* out_onehot = out_lv + (size_t)N * VQ_INDIM;
    float* out_loss   = out_onehot + (size_t)N * K;

    // workspace carve-up (all offsets 256B-aligned); total ~60.3 MB
    char* ws = (char*)d_ws;
    size_t off = 0;
    auto carve = [&](size_t bytes) { char* p = ws + off; off += (bytes + 255) & ~(size_t)255; return p; };
    f16*   x16   = (f16*)  carve((size_t)N * 128 * 2);
    f16*   h1    = (f16*)  carve((size_t)N * 256 * 2);   // enc h1, reused as dec h2
    f16*   h2    = (f16*)  carve((size_t)N * 128 * 2);   // enc h2, reused as dec h1
    f16*   lat16 = (f16*)  carve((size_t)N * 64 * 2);
    float* lat32 = (float*)carve((size_t)N * 64 * 4);
    f16*   q16   = (f16*)  carve((size_t)N * 64 * 2);
    float* q32   = (float*)carve((size_t)N * 64 * 4);
    f16*   emb16 = (f16*)  carve((size_t)K * 64 * 2);
    float* enorm = (float*)carve((size_t)K * 4);
    f16*   w1    = (f16*)  carve(256 * 128 * 2);
    f16*   w2    = (f16*)  carve(128 * 256 * 2);
    f16*   w3    = (f16*)  carve(64 * 128 * 2);
    f16*   dw1   = (f16*)  carve(128 * 64 * 2);
    f16*   dw2   = (f16*)  carve(256 * 128 * 2);
    f16*   wmu   = (f16*)  carve(128 * 256 * 2);
    f16*   wlv   = (f16*)  carve(128 * 256 * 2);
    int*   inds  = (int*)  carve((size_t)N * 4);

    // --- conversions ---
    auto cvt = [&](const float* src, f16* dst, int n) {
        f32_to_f16_kernel<<<(n + 255) / 256, 256, 0, stream>>>(src, dst, n);
    };
    cvt(x,    x16,   N * 128);
    cvt(eW1,  w1,    256 * 128);
    cvt(eW2,  w2,    128 * 256);
    cvt(eW3,  w3,    64 * 128);
    cvt(emb,  emb16, K * 64);
    cvt(dW1,  dw1,   128 * 64);
    cvt(dW2,  dw2,   256 * 128);
    cvt(dWmu, wmu,   128 * 256);
    cvt(dWlv, wlv,   128 * 256);
    emb_halfnorm_kernel<<<(K + 255) / 256, 256, 0, stream>>>(emb, enorm, K);

    // zero one_hot (1.07 GB of b128 stores) + loss slot
    fill_zero_kernel<<<2048, 256, 0, stream>>>(out_onehot, (size_t)N * K / 4, out_loss);

    // --- encoder ---
    const int mt = N / 16;  // 2048 row tiles
    gemm_wmma_kernel<<<dim3(mt, 4), 128, 0, stream>>>(x16, w1, eb1, nullptr, h1,    N, 128, 256, 1);
    gemm_wmma_kernel<<<dim3(mt, 2), 128, 0, stream>>>(h1,  w2, eb2, nullptr, h2,    N, 256, 128, 1);
    gemm_wmma_kernel<<<dim3(mt, 1), 128, 0, stream>>>(h2,  w3, eb3, lat32,   lat16, N, 128, 64,  0);

    // --- vector quantizer ---
    vq_argmax_kernel<<<N / 128, 256, 0, stream>>>(lat16, emb16, enorm, inds, N, K);
    vq_post_kernel<<<(N + 255) / 256, 256, 0, stream>>>(inds, emb, emb16, lat32,
                                                        q32, q16, out_onehot, out_loss, N, K);

    // --- decoder (z == q forward) ---
    gemm_wmma_kernel<<<dim3(mt, 2), 128, 0, stream>>>(q16, dw1, db1,  nullptr, h2, N, 64,  128, 1);
    gemm_wmma_kernel<<<dim3(mt, 4), 128, 0, stream>>>(h2,  dw2, db2,  nullptr, h1, N, 128, 256, 1);
    gemm_wmma_kernel<<<dim3(mt, 2), 128, 0, stream>>>(h1,  wmu, dbmu, out_mu, nullptr, N, 256, 128, 0);
    gemm_wmma_kernel<<<dim3(mt, 2), 128, 0, stream>>>(h1,  wlv, dblv, out_lv, nullptr, N, 256, 128, 0);
}
